// CapInfoNCE_61435212202616
// MI455X (gfx1250) — compile-verified, matrix operand order checked by reading, not run on hardware
//
#include <hip/hip_runtime.h>

// CapInfoNCE fused loss for MI455X (gfx1250), wave32 + WMMA bf16 (split hi/lo x3)
// + Tensor Data Mover (TDM) staging of the W tile into LDS with hardware row padding.
//
// B=96 (both batches), To=50, Tw=40, D=512.
// Stage 0: split f32 -> bf16 hi + bf16 lo (error compensation term).
// Stage 1: per (w,b) pair: S = W(40x512) x O^T(512x50) via v_wmma_f32_16x16x32_bf16,
//          3 WMMAs per K-step (hi*hi + hi*lo + lo*hi) ~ f32 accuracy,
//          W tile staged by TENSOR_LOAD_TO_LDS (pad 4 DWORDs per 256 -> stride 520),
//          then fused softmax-weighted self-sum -> logits[w][t][b].
// Stage 2: per (w,t): LSE over b, diag term; Stage 3: deterministic tree sum.

#define BATCH 96
#define TO    50
#define TW    40
#define DIM   512
#define LDSW  520           // LDS row stride in bf16 elems (16B aligned, conflict-free)
#define SSTR  65            // score LDS row stride in floats

typedef __attribute__((ext_vector_type(16))) __bf16 v16bf;
typedef __attribute__((ext_vector_type(8)))  float  v8f;
typedef unsigned int u32x4 __attribute__((ext_vector_type(4)));
typedef int          i32x4 __attribute__((ext_vector_type(4)));
typedef int          i32x8 __attribute__((ext_vector_type(8)));

static __device__ __forceinline__ v16bf frag16(const __bf16* p) {
    union { uint4 u[2]; v16bf v; } f;
    f.u[0] = *(const uint4*)(p);
    f.u[1] = *(const uint4*)(p + 16);
    return f.v;
}

#define WMMA_BF16(A, B, C) \
    __builtin_amdgcn_wmma_f32_16x16x32_bf16(false, (A), false, (B), (short)0, (C), false, false)

// Issue one TDM 2D tile load: 40 rows x 512 bf16, row-contiguous in global,
// written to LDS with 16B padding after every 1024B row (=> 520-elem stride).
// D# bit layout per CDNA5 ISA.
static __device__ __forceinline__ void tdm_load_w_tile(unsigned int lds_off, const void* gptr) {
    unsigned long long ga = (unsigned long long)(uintptr_t)gptr;

    u32x4 g0;
    g0[0] = 1u;                                   // count=1, user descriptor, no gather
    g0[1] = lds_off;                              // lds_addr (bytes, workgroup-relative)
    g0[2] = (unsigned int)ga;                     // global_addr[31:0]
    g0[3] = (unsigned int)((ga >> 32) & 0x01FFFFFFu) | (2u << 30);  // addr[56:32] | type=2

    i32x8 g1;
    g1[0] = (int)((1u << 16)      // data_size = 1 -> 2 bytes
                | (1u << 20)      // pad_enable
                | (7u << 22)      // pad_interval = 7 -> every 256 DWORDs (1024B = one row)
                | (3u << 25));    // pad_amount = 3 -> 4 DWORDs (16B)  => stride 1040B
    g1[1] = (int)(((unsigned)DIM & 0xFFFFu) << 16);   // tensor_dim0[15:0] = 512
    g1[2] = (int)(((unsigned)TW) << 16);              // tensor_dim0 hi=0 | tensor_dim1[15:0]=40
    g1[3] = (int)(((unsigned)DIM) << 16);             // tensor_dim1 hi=0 | tile_dim0 = 512
    g1[4] = (int)TW;                                  // tile_dim1 = 40, tile_dim2 = 0
    g1[5] = (int)DIM;                                 // tensor_dim0_stride = 512 (low 32)
    g1[6] = 0;                                        // stride hi | tensor_dim1_stride lo
    g1[7] = 0;

    i32x4 z4 = {0, 0, 0, 0};
#if defined(__clang_major__) && (__clang_major__ >= 23)
    i32x8 z8 = {0, 0, 0, 0, 0, 0, 0, 0};
    __builtin_amdgcn_tensor_load_to_lds(g0, g1, z4, z4, z8, 0);
#else
    __builtin_amdgcn_tensor_load_to_lds(g0, g1, z4, z4, 0);
#endif
}

// ---------------- Stage 0: f32 -> bf16 hi/lo split ----------------
__global__ __launch_bounds__(256) void split_bf16_kernel(const float* __restrict__ x,
                                                         __bf16* __restrict__ hi,
                                                         __bf16* __restrict__ lo, int n) {
    int i = blockIdx.x * blockDim.x + threadIdx.x;
    if (i < n) {
        float v = x[i];
        __bf16 h = (__bf16)v;
        hi[i] = h;
        lo[i] = (__bf16)(v - (float)h);
    }
}

// ---------------- Stage 1: cross-batch GEMM + fused softmax ----------------
__global__ __launch_bounds__(128) void attn_logits_kernel(const __bf16* __restrict__ Whi,
                                                          const __bf16* __restrict__ Wlo,
                                                          const __bf16* __restrict__ Ohi,
                                                          const __bf16* __restrict__ Olo,
                                                          float* __restrict__ logits) {
    const int bb = blockIdx.x;   // o-batch index
    const int wb = blockIdx.y;   // w-batch index

    extern __shared__ char smem[];
    __bf16* sWhi = (__bf16*)smem;              // 48 x LDSW
    __bf16* sWlo = sWhi + 48 * LDSW;           // 48 x LDSW
    float*  sS   = (float*)(sWlo + 48 * LDSW); // 48 x SSTR

    const int tid = threadIdx.x;

    const __bf16* gWhi = Whi + (size_t)wb * TW * DIM;
    const __bf16* gWlo = Wlo + (size_t)wb * TW * DIM;

    // Zero M-padding rows 40..47 (cols 0..511 are all fragments ever touch).
    for (int i = tid; i < 8 * 64; i += 128) {
        int r = TW + (i >> 6);
        int c = (i & 63) * 8;
        uint4 z = make_uint4(0u, 0u, 0u, 0u);
        *(uint4*)(sWhi + r * LDSW + c) = z;
        *(uint4*)(sWlo + r * LDSW + c) = z;
    }

    // Wave 0 kicks the TDM for both W tiles and waits for completion.
    if (tid < 32) {
        tdm_load_w_tile((unsigned int)(uintptr_t)sWhi, gWhi);
        tdm_load_w_tile((unsigned int)(uintptr_t)sWlo, gWlo);
        __builtin_amdgcn_s_wait_tensorcnt(0);
    }
    __syncthreads();

    const int wave = tid >> 5;
    const int lane = tid & 31;
    const int fr   = lane & 15;           // fragment row (M for A, N for B)
    const int kb   = (lane >> 4) * 8;     // per-lane K base within 32-wide K slice
    const int nt   = wave;                // each wave owns one N-tile (16 O rows)

    int rowO = nt * 16 + fr;
    if (rowO > TO - 1) rowO = TO - 1;     // clamp padded rows (values discarded later)
    const __bf16* gOhi = Ohi + ((size_t)bb * TO + rowO) * DIM;
    const __bf16* gOlo = Olo + ((size_t)bb * TO + rowO) * DIM;

    v8f acc0 = {}, acc1 = {}, acc2 = {};

    for (int ks = 0; ks < DIM / 32; ++ks) {
        const int k = ks * 32 + kb;

        v16bf bh = frag16(gOhi + k);
        v16bf bl = frag16(gOlo + k);

        v16bf a0h = frag16(sWhi + (0 * 16 + fr) * LDSW + k);
        v16bf a1h = frag16(sWhi + (1 * 16 + fr) * LDSW + k);
        v16bf a2h = frag16(sWhi + (2 * 16 + fr) * LDSW + k);
        v16bf a0l = frag16(sWlo + (0 * 16 + fr) * LDSW + k);
        v16bf a1l = frag16(sWlo + (1 * 16 + fr) * LDSW + k);
        v16bf a2l = frag16(sWlo + (2 * 16 + fr) * LDSW + k);

        acc0 = WMMA_BF16(a0h, bh, acc0);
        acc1 = WMMA_BF16(a1h, bh, acc1);
        acc2 = WMMA_BF16(a2h, bh, acc2);
        acc0 = WMMA_BF16(a0h, bl, acc0);
        acc1 = WMMA_BF16(a1h, bl, acc1);
        acc2 = WMMA_BF16(a2h, bl, acc2);
        acc0 = WMMA_BF16(a0l, bh, acc0);
        acc1 = WMMA_BF16(a1l, bh, acc1);
        acc2 = WMMA_BF16(a2l, bh, acc2);
    }

    // Scatter accumulators into score LDS (C/D layout: m = v + 8*(lane>>4), n = lane&15).
    const int orow = (lane >> 4) * 8;
    const int ocol = nt * 16 + (lane & 15);
#pragma unroll
    for (int v = 0; v < 8; ++v) {
        sS[(0 * 16 + orow + v) * SSTR + ocol] = acc0[v];
        sS[(1 * 16 + orow + v) * SSTR + ocol] = acc1[v];
        sS[(2 * 16 + orow + v) * SSTR + ocol] = acc2[v];
    }
    __syncthreads();

    // Fused: logits[w,b,t] = sum_o softmax_o(scale*S) * S  (one thread per t row).
    if (tid < TW) {
        const float scale = 0.04419417382415922f;   // 1/sqrt(512)
        const float* row = sS + tid * SSTR;
        float mx = -3.0e38f;
        for (int i = 0; i < TO; ++i) mx = fmaxf(mx, row[i] * scale);
        float se = 0.0f, wsum = 0.0f;
        for (int i = 0; i < TO; ++i) {
            float s = row[i];
            float e = __expf(s * scale - mx);
            se += e;
            wsum += e * s;
        }
        logits[((size_t)wb * TW + tid) * BATCH + bb] = wsum / se;
    }
}

// ---------------- Stage 2: per-(w,t) LSE over b + diag ----------------
__global__ __launch_bounds__(32) void lse_kernel(const float* __restrict__ logits,
                                                 float* __restrict__ partial) {
    const int id = blockIdx.x;          // 0 .. 96*40-1
    const int w = id / TW;
    const float* row = logits + (size_t)id * BATCH;   // logits[w][t][:]
    const int lane = threadIdx.x;

    float a = row[lane], b = row[lane + 32], c = row[lane + 64];
    float m = fmaxf(a, fmaxf(b, c));
#pragma unroll
    for (int off = 16; off > 0; off >>= 1) m = fmaxf(m, __shfl_xor(m, off, 32));
    float s = __expf(a - m) + __expf(b - m) + __expf(c - m);
#pragma unroll
    for (int off = 16; off > 0; off >>= 1) s += __shfl_xor(s, off, 32);
    if (lane == 0) partial[id] = row[w] - (m + __logf(s));   // log_softmax diag term
}

// ---------------- Stage 3: deterministic final reduction ----------------
__global__ __launch_bounds__(256) void finalize_kernel(const float* __restrict__ partial,
                                                       float* __restrict__ out) {
    __shared__ float sm[256];
    float acc = 0.0f;
    for (int i = threadIdx.x; i < BATCH * TW; i += 256) acc += partial[i];
    sm[threadIdx.x] = acc;
    __syncthreads();
    for (int s = 128; s > 0; s >>= 1) {
        if (threadIdx.x < s) sm[threadIdx.x] += sm[threadIdx.x + s];
        __syncthreads();
    }
    if (threadIdx.x == 0) out[0] = -sm[0] / (float)(BATCH * TW);
}

extern "C" void kernel_launch(void* const* d_in, const int* in_sizes, int n_in,
                              void* d_out, int out_size, void* d_ws, size_t ws_size,
                              hipStream_t stream) {
    const float* o = (const float*)d_in[0];   // (96, 50, 512)
    const float* w = (const float*)d_in[1];   // (96, 40, 512)
    float* out = (float*)d_out;

    const int nW = BATCH * TW * DIM;          // 1,966,080
    const int nO = BATCH * TO * DIM;          // 2,457,600

    // Workspace layout (all 16B aligned)
    char* ws = (char*)d_ws;
    __bf16* whi = (__bf16*)(ws);
    __bf16* wlo = (__bf16*)(ws + (size_t)nW * 2);
    __bf16* ohi = (__bf16*)(ws + (size_t)nW * 4);
    __bf16* olo = (__bf16*)(ws + (size_t)nW * 4 + (size_t)nO * 2);
    float*  logits  = (float*)(ws + (size_t)nW * 4 + (size_t)nO * 4);
    float*  partial = logits + (size_t)BATCH * TW * BATCH;

    // Stage 0: split f32 inputs into bf16 hi + lo
    split_bf16_kernel<<<(nW + 255) / 256, 256, 0, stream>>>(w, whi, wlo, nW);
    split_bf16_kernel<<<(nO + 255) / 256, 256, 0, stream>>>(o, ohi, olo, nO);

    // Stage 1: 96x96 cross-batch tiles, TDM-staged W + WMMA GEMM + fused softmax
    {
        dim3 grid(BATCH, BATCH);
        size_t smem = (size_t)(2 * 48 * LDSW) * sizeof(__bf16) + (size_t)(48 * SSTR) * sizeof(float);
        attn_logits_kernel<<<grid, 128, smem, stream>>>(whi, wlo, ohi, olo, logits);
    }

    // Stage 2: LSE over b for each (w,t); diag - LSE partials
    lse_kernel<<<BATCH * TW, 32, 0, stream>>>(logits, partial);

    // Stage 3: deterministic tree reduction -> scalar loss
    finalize_kernel<<<1, 256, 0, stream>>>(partial, out);
}
